// MessagePassingNeuralNetwork_64424509440354
// MI455X (gfx1250) — matrix operand order repaired and checked by reading, not compiled, not run on hardware
//
#include <hip/hip_runtime.h>
#include <hip/hip_bf16.h>

typedef __attribute__((ext_vector_type(16))) __bf16 v16bf;
typedef __attribute__((ext_vector_type(8)))  float  v8f;
typedef unsigned short u16;
typedef unsigned int   u32;

#define N_NODES 20000
#define N_EDGES 640000
#define DDIM    128
#define HDIM    256

// f32 -> bf16 round-to-nearest-even
__device__ __forceinline__ u16 f2bf(float f) {
    u32 u = __float_as_uint(f);
    u += 0x7FFFu + ((u >> 16) & 1u);
    return (u16)(u >> 16);
}

__device__ __forceinline__ void store4bf(u16* p, float4 v) {
    uint2 r;
    r.x = (u32)f2bf(v.x) | ((u32)f2bf(v.y) << 16);
    r.y = (u32)f2bf(v.z) | ((u32)f2bf(v.w) << 16);
    *(uint2*)p = r;
}

// Build a 16-element bf16 fragment from two 16B LDS chunks
__device__ __forceinline__ v16bf mkfrag(const u16* p0, const u16* p1) {
    union { uint4 u[2]; v16bf v; } x;
    x.u[0] = *(const uint4*)p0;
    x.u[1] = *(const uint4*)p1;
    return x.v;
}

// Async 16B copy global -> LDS (CDNA5 GLOBAL_LOAD_ASYNC_TO_LDS_B128, ASYNCcnt)
__device__ __forceinline__ void async_cp16(u16* lds_p, const u16* g_p) {
    u32 lds_off = (u32)(size_t)(void*)lds_p;   // LDS aperture: addr[31:0] = LDS offset
    asm volatile("global_load_async_to_lds_b128 %0, %1, off"
                 :: "v"(lds_off), "v"(g_p) : "memory");
}

__device__ __forceinline__ void wait_async0() {
    asm volatile("s_wait_asynccnt 0x0" ::: "memory");
}

// Generic tiled GEMM: C[strip 16 x NTILES*16] += A[strip, K] * W^T
//  As : bf16 LDS, row-major, pitch `apitch` (bf16 units), K = KSTEPS*32
//  gW : bf16 global, layout [nrowsB][gK] (W transposed: row = output col)
//  Bs0/Bs1 : double-buffered LDS staging [nrowsB][40], filled by async DMA
template<int KSTEPS, int NTILES>
__device__ __forceinline__ void gemm_tiles(
    const u16* __restrict__ As, int apitch,
    u16* __restrict__ Bs0, u16* __restrict__ Bs1,
    const u16* __restrict__ gW, int gK, int nrowsB,
    v8f* acc, int mstrip, int tid)
{
    const int lane = tid & 31;
    const int hh   = lane >> 4;   // half-wave selector
    const int ln   = lane & 15;

    auto stage = [&](int ks, u16* Bs) {
        const int nchunk = nrowsB * 4;       // 16B chunks
        for (int c = tid; c < nchunk; c += 256) {
            int row = c >> 2, kc = c & 3;
            async_cp16(Bs + row * 40 + kc * 8,
                       gW + row * gK + ks * 32 + kc * 8);
        }
    };

    stage(0, Bs0);                            // prologue prefetch
    for (int ks = 0; ks < KSTEPS; ++ks) {
        u16* cur = (ks & 1) ? Bs1 : Bs0;
        u16* nxt = (ks & 1) ? Bs0 : Bs1;
        wait_async0();                        // own slab-`ks` loads done
        __syncthreads();                      // => everyone's slab-`ks` loads done
        if (ks + 1 < KSTEPS) stage(ks + 1, nxt);  // overlap next fetch w/ compute

        // A fragment: lane<16 holds K{0..15}, lane>=16 holds K{16..31}
        const u16* arow = As + (mstrip * 16 + ln) * apitch + ks * 32 + hh * 16;
        v16bf a = mkfrag(arow, arow + 8);

        // software-pipelined B fragments: load nt+1 while WMMA nt runs
        const u16* b0 = cur + ln * 40 + hh * 16;
        v16bf bcur = mkfrag(b0, b0 + 8);
        #pragma unroll
        for (int nt = 0; nt < NTILES; ++nt) {
            v16bf bnext = bcur;
            if (nt + 1 < NTILES) {
                const u16* bp = cur + ((nt + 1) * 16 + ln) * 40 + hh * 16;
                bnext = mkfrag(bp, bp + 8);
            }
            acc[nt] = __builtin_amdgcn_wmma_f32_16x16x32_bf16(
                false, a, false, bnext == bnext ? bcur : bcur, (short)0, acc[nt],
                false, false);
            bcur = bnext;
        }
    }
}

// ---------------- weight transpose + bf16 convert:  out[n][k] = in[k][n] -----
__global__ void xpose_bf16(const float* __restrict__ in, u16* __restrict__ out,
                           int K, int N) {
    int i = blockIdx.x * 256 + threadIdx.x;
    if (i >= K * N) return;
    int n = i / K, k = i - n * K;
    out[i] = f2bf(in[(size_t)k * N + n]);
}

// ---------------- edge kernel: 128 edges / block ----------------------------
#define EPITCH 392   // 384 + 8 bf16 pad -> 196 dwords, conflict-free
#define HPITCH 264   // 256 + 8 bf16 pad -> 132 dwords, conflict-free

__global__ __launch_bounds__(256) void edge_kernel(
    const float* __restrict__ nf, const float* __restrict__ ef,
    const int* __restrict__ src, const int* __restrict__ dst,
    const u16* __restrict__ We1T, const float* __restrict__ be1,
    const u16* __restrict__ We2T, const float* __restrict__ be2,
    float* __restrict__ outEf, float* __restrict__ agg)
{
    __shared__ __align__(16) u16 As[128 * EPITCH];   // A tile, reused as hidden
    __shared__ __align__(16) u16 Bs0[256 * 40];      // async-filled weight slabs
    __shared__ __align__(16) u16 Bs1[256 * 40];
    __shared__ int srcS[128];
    __shared__ int dstS[128];

    const int tid    = threadIdx.x;
    const int lane   = tid & 31;
    const int mstrip = tid >> 5;         // 8 waves -> 8 strips of 16 edges
    const int base   = blockIdx.x * 128;
    const int hh = lane >> 4, ln = lane & 15;

    if (tid < 128) { srcS[tid] = src[base + tid]; dstS[tid] = dst[base + tid]; }
    __syncthreads();

    // stage [ef | nf[src] | nf[dst]] as bf16 rows of 384
    #pragma unroll 4
    for (int i = 0; i < 48; ++i) {
        int c   = tid + 256 * i;         // 16B-chunk id, 96 chunks per row
        int row = c / 96;
        int q   = c - row * 96;
        int seg = q >> 5, qi = q & 31;
        const float4* p;
        if (seg == 0)      p = (const float4*)(ef + (size_t)(base + row) * DDIM) + qi;
        else if (seg == 1) p = (const float4*)(nf + (size_t)srcS[row] * DDIM) + qi;
        else               p = (const float4*)(nf + (size_t)dstS[row] * DDIM) + qi;
        store4bf(As + row * EPITCH + seg * 128 + qi * 4, *p);
    }

    // GEMM1: [128,384] x [384,256]
    v8f acc[16];
    #pragma unroll
    for (int t = 0; t < 16; ++t) acc[t] = (v8f){0,0,0,0,0,0,0,0};
    gemm_tiles<12, 16>(As, EPITCH, Bs0, Bs1, We1T, 384, 256, acc, mstrip, tid);

    __syncthreads();   // everyone done reading A-layout before hidden overwrite

    // bias + ReLU -> hidden bf16 tile (reuse As region, per-strip ownership)
    #pragma unroll
    for (int nt = 0; nt < 16; ++nt) {
        float b1 = be1[nt * 16 + ln];
        #pragma unroll
        for (int j = 0; j < 8; ++j) {
            int m = mstrip * 16 + hh * 8 + j;
            float v = acc[nt][j] + b1;
            As[m * HPITCH + nt * 16 + ln] = f2bf(v > 0.f ? v : 0.f);
        }
    }

    // GEMM2: [128,256] x [256,128]
    v8f acc2[8];
    #pragma unroll
    for (int t = 0; t < 8; ++t) acc2[t] = (v8f){0,0,0,0,0,0,0,0};
    gemm_tiles<8, 8>(As, HPITCH, Bs0, Bs1, We2T, 256, 128, acc2, mstrip, tid);

    // epilogue: updated_ef store + segment-sum atomics at dst
    #pragma unroll
    for (int nt = 0; nt < 8; ++nt) {
        int n = nt * 16 + ln;
        float b2 = be2[n];
        #pragma unroll
        for (int j = 0; j < 8; ++j) {
            int m = mstrip * 16 + hh * 8 + j;
            float v = acc2[nt][j] + b2;
            outEf[(size_t)(base + m) * DDIM + n] = v;
            atomicAdd(agg + (size_t)dstS[m] * DDIM + n, v);
        }
    }
}

// ---------------- node kernel: 128 nodes / block ----------------------------
__global__ __launch_bounds__(256) void node_kernel(
    const float* __restrict__ nf, const float* __restrict__ agg,
    const u16* __restrict__ Wn1T, const float* __restrict__ bn1,
    const u16* __restrict__ Wn2T, const float* __restrict__ bn2,
    float* __restrict__ outNf)
{
    __shared__ __align__(16) u16 As[128 * HPITCH];   // K = 256, reused as hidden
    __shared__ __align__(16) u16 Bs0[256 * 40];
    __shared__ __align__(16) u16 Bs1[256 * 40];

    const int tid    = threadIdx.x;
    const int lane   = tid & 31;
    const int mstrip = tid >> 5;
    const int base   = blockIdx.x * 128;
    const int hh = lane >> 4, ln = lane & 15;

    // stage [agg | nf] rows of 256 (zero-fill past N_NODES)
    #pragma unroll 4
    for (int i = 0; i < 32; ++i) {
        int c   = tid + 256 * i;          // 64 chunks per row
        int row = c >> 6;
        int q   = c & 63;
        int seg = q >> 5, qi = q & 31;
        int node = base + row;
        float4 v = make_float4(0.f, 0.f, 0.f, 0.f);
        if (node < N_NODES) {
            const float* sp = (seg == 0) ? (agg + (size_t)node * DDIM)
                                         : (nf  + (size_t)node * DDIM);
            v = ((const float4*)sp)[qi];
        }
        store4bf(As + row * HPITCH + seg * 128 + qi * 4, v);
    }

    v8f acc[16];
    #pragma unroll
    for (int t = 0; t < 16; ++t) acc[t] = (v8f){0,0,0,0,0,0,0,0};
    gemm_tiles<8, 16>(As, HPITCH, Bs0, Bs1, Wn1T, 256, 256, acc, mstrip, tid);

    __syncthreads();

    #pragma unroll
    for (int nt = 0; nt < 16; ++nt) {
        float b1 = bn1[nt * 16 + ln];
        #pragma unroll
        for (int j = 0; j < 8; ++j) {
            int m = mstrip * 16 + hh * 8 + j;
            float v = acc[nt][j] + b1;
            As[m * HPITCH + nt * 16 + ln] = f2bf(v > 0.f ? v : 0.f);
        }
    }

    v8f acc2[8];
    #pragma unroll
    for (int t = 0; t < 8; ++t) acc2[t] = (v8f){0,0,0,0,0,0,0,0};
    gemm_tiles<8, 8>(As, HPITCH, Bs0, Bs1, Wn2T, 256, 128, acc2, mstrip, tid);

    #pragma unroll
    for (int nt = 0; nt < 8; ++nt) {
        int n = nt * 16 + ln;
        float b2 = bn2[n];
        #pragma unroll
        for (int j = 0; j < 8; ++j) {
            int m = mstrip * 16 + hh * 8 + j;
            int node = base + m;
            if (node < N_NODES)
                outNf[(size_t)node * DDIM + n] = acc2[nt][j] + b2;
        }
    }
}

extern "C" void kernel_launch(void* const* d_in, const int* in_sizes, int n_in,
                              void* d_out, int out_size, void* d_ws, size_t ws_size,
                              hipStream_t stream) {
    const float* nf  = (const float*)d_in[0];
    const float* ef  = (const float*)d_in[1];
    const int*   src = (const int*)d_in[2];
    const int*   dst = (const int*)d_in[3];
    const float* We1 = (const float*)d_in[4];
    const float* be1 = (const float*)d_in[5];
    const float* We2 = (const float*)d_in[6];
    const float* be2 = (const float*)d_in[7];
    const float* Wn1 = (const float*)d_in[8];
    const float* bn1 = (const float*)d_in[9];
    const float* Wn2 = (const float*)d_in[10];
    const float* bn2 = (const float*)d_in[11];

    float* outNf = (float*)d_out;                       // [20000,128]
    float* outEf = outNf + (size_t)N_NODES * DDIM;      // [640000,128]

    char* ws = (char*)d_ws;
    float* agg = (float*)ws;                                  // 20000*128 f32
    u16* We1T = (u16*)(ws + (size_t)N_NODES * DDIM * 4);      // [256][384]
    u16* We2T = We1T + 256 * 384;                             // [128][256]
    u16* Wn1T = We2T + 128 * 256;                             // [256][256]
    u16* Wn2T = Wn1T + 256 * 256;                             // [128][256]

    hipMemsetAsync(agg, 0, (size_t)N_NODES * DDIM * sizeof(float), stream);
    xpose_bf16<<<(384 * 256 + 255) / 256, 256, 0, stream>>>(We1, We1T, 384, 256);
    xpose_bf16<<<(256 * 128 + 255) / 256, 256, 0, stream>>>(We2, We2T, 256, 128);
    xpose_bf16<<<(256 * 256 + 255) / 256, 256, 0, stream>>>(Wn1, Wn1T, 256, 256);
    xpose_bf16<<<(256 * 128 + 255) / 256, 256, 0, stream>>>(Wn2, Wn2T, 256, 128);

    edge_kernel<<<N_EDGES / 128, 256, 0, stream>>>(
        nf, ef, src, dst, We1T, be1, We2T, be2, outEf, agg);
    node_kernel<<<(N_NODES + 127) / 128, 256, 0, stream>>>(
        nf, agg, Wn1T, bn1, Wn2T, bn2, outNf);
}
